// WeatherPrediction_22969485099524
// MI455X (gfx1250) — matrix operand order repaired and checked by reading, not compiled, not run on hardware
//
#include <hip/hip_runtime.h>
#include <cstddef>

// ---------------- model constants ----------------
#define DLAT   256
#define NSP    65160      // 181*360 spatial points
#define NSPH   2883       // sphere nodes
#define NEDGE  17298      // NSPH*6 edges
#define FDIM   78

// ---------------- WMMA types ----------------
typedef __attribute__((ext_vector_type(16))) __bf16   v16bf;
typedef __attribute__((ext_vector_type(8)))  float    v8f;
typedef __attribute__((ext_vector_type(4)))  unsigned v4u;

struct V16 { v4u lo, hi; };   // 32B, bit_cast -> v16bf

__device__ __forceinline__ unsigned f2bf(float f) {
  // round-to-nearest-even f32 -> bf16 (as low 16 bits)
  unsigned u = __builtin_bit_cast(unsigned, f);
  u += 0x7FFFu + ((u >> 16) & 1u);
  return u >> 16;
}
__device__ __forceinline__ unsigned pack2(float lo, float hi) {
  return f2bf(lo) | (f2bf(hi) << 16);
}

// =====================================================================
// Tiled bf16-WMMA GEMM:  C[M,N] (=|+=) gatherA(A)[M,K] @ W[K,N] (+bias)
//   A row index = idx ? idx[m] : m   (gather fused into tile staging)
//   flags: bit0 = accumulate into existing C, bit1 = relu at the end
// 64x128 workgroup tile, 8 waves; each wave: 1 A frag x 4 B frags
//   -> 4x v_wmma_f32_16x16x32_bf16 per k-step.
// LDS tiles stored as packed bf16x2 words, fragment-order, 80B row
// stride (16B-aligned rows; 20-word stride => conflict-free b128 reads).
// =====================================================================
#define TM 64
#define TN 128
#define TK 32

__global__ __launch_bounds__(256) void gemm_wmma(
    const float* __restrict__ A, int lda,
    const int*   __restrict__ idx,
    const float* __restrict__ W, int ldw,
    const float* __restrict__ bias,
    float* __restrict__ C, int ldc,
    int M, int N, int K, int flags)
{
  __shared__ __align__(16) unsigned Asm[TM][20];  // [row][kpair], 16 used + 4 pad
  __shared__ __align__(16) unsigned Bsm[TN][20];  // transposed: [col][kpair]

  const int t    = threadIdx.x;
  const int lane = t & 31;
  const int wid  = t >> 5;
  const int msub = (wid & 3) << 4;        // 0,16,32,48
  const int nofs = (wid >> 2) << 6;       // 0,64
  const int half = lane >> 4;
  const int mn   = lane & 15;
  const int tileM = blockIdx.y * TM;
  const int tileN = blockIdx.x * TN;

  // ---- hoist gathered A row pointers (4 rows per thread, branch-free k-loop)
  const float* aptr[4];
#pragma unroll
  for (int i = 0; i < 4; ++i) {
    int gm  = tileM + (t >> 4) + (i << 4);
    int gmc = gm < M ? gm : (M - 1);
    int ar  = idx ? idx[gmc] : gmc;
    aptr[i] = A + (size_t)ar * lda;
  }
  const int kpA = t & 15;                 // k-pair staged by this thread (A)
  const int nB  = t & 127;                // column staged by this thread (B)
  const int gnB = tileN + nB;
  const int nBc = gnB < N ? gnB : (N - 1);

  v8f cacc[4];
#pragma unroll
  for (int j = 0; j < 4; ++j) cacc[j] = (v8f){0.f,0.f,0.f,0.f,0.f,0.f,0.f,0.f};

  const int ktiles = (K + TK - 1) / TK;
  for (int kt = 0; kt < ktiles; ++kt) {
    const int kb = kt * TK;

    // ---- stage A (64x32) as packed pairs: clamped loads + select-zero ----
    {
      const int c0 = kb + 2 * kpA, c1 = c0 + 1;
      const int c0c = c0 < K ? c0 : K - 1;
      const int c1c = c1 < K ? c1 : K - 1;
#pragma unroll
      for (int i = 0; i < 4; ++i) {
        float v0 = aptr[i][c0c];
        float v1 = aptr[i][c1c];
        v0 = (c0 < K) ? v0 : 0.f;
        v1 = (c1 < K) ? v1 : 0.f;
        Asm[(t >> 4) + (i << 4)][kpA] = pack2(v0, v1);
      }
    }
    // ---- stage B (32x128) transposed to [col][kpair] ----
#pragma unroll
    for (int i = 0; i < 8; ++i) {
      int kp = (t >> 7) + (i << 1);
      int k0 = kb + 2 * kp, k1 = k0 + 1;
      int k0c = k0 < K ? k0 : K - 1;
      int k1c = k1 < K ? k1 : K - 1;
      float v0 = W[(size_t)k0c * ldw + nBc];
      float v1 = W[(size_t)k1c * ldw + nBc];
      v0 = (k0 < K) ? v0 : 0.f;
      v1 = (k1 < K) ? v1 : 0.f;
      Bsm[nB][kp] = pack2(v0, v1);
    }
    __syncthreads();

    // ---- fragments via wide LDS loads (2x ds_load_b128 each) ----
    const v4u* pa = (const v4u*)&Asm[msub + mn][0];
    V16 av; av.lo = pa[half]; av.hi = pa[2 + half];
    v16bf a = __builtin_bit_cast(v16bf, av);
#pragma unroll
    for (int j = 0; j < 4; ++j) {
      const v4u* pb = (const v4u*)&Bsm[nofs + (j << 4) + mn][0];
      V16 bv; bv.lo = pb[half]; bv.hi = pb[2 + half];
      v16bf b = __builtin_bit_cast(v16bf, bv);
      cacc[j] = __builtin_amdgcn_wmma_f32_16x16x32_bf16(
          false, a, false, b, (short)0, cacc[j], false, false);
    }
    __syncthreads();
  }

  // ---- epilogue: C layout = row (vgpr + 8*half), col (lane&15) ----
#pragma unroll
  for (int j = 0; j < 4; ++j) {
    const int gn = tileN + nofs + (j << 4) + mn;
    if (gn < N) {
      const float bv = bias ? bias[gn] : 0.f;
#pragma unroll
      for (int r = 0; r < 8; ++r) {
        const int gm = tileM + msub + r + (half << 3);
        if (gm < M) {
          float v = cacc[j][r] + bv;
          size_t o = (size_t)gm * ldc + gn;
          if (flags & 1) v += C[o];
          if (flags & 2) v = fmaxf(v, 0.f);
          C[o] = v;
        }
      }
    }
  }
}

// ---------------- per-edge ||sphere[si]-spatial[pj]|| (one wave / edge) ---
__global__ __launch_bounds__(256) void edge_d3(
    const float* __restrict__ sph, const float* __restrict__ spt,
    const int* __restrict__ si, const int* __restrict__ pj,
    float* __restrict__ d3, int E)
{
  int e = blockIdx.x * 8 + (threadIdx.x >> 5);
  if (e >= E) return;
  int lane = threadIdx.x & 31;
  const float* a = sph + (size_t)si[e] * DLAT;
  const float* b = spt + (size_t)pj[e] * DLAT;
  float acc = 0.f;
#pragma unroll
  for (int j = 0; j < 8; ++j) {
    float d = a[lane + 32 * j] - b[lane + 32 * j];
    acc += d * d;
  }
  for (int off = 16; off > 0; off >>= 1) acc += __shfl_xor(acc, off, 32);
  if (lane == 0) d3[e] = sqrtf(acc);
}

// --------- H[e,n] = b[n] + dlat*W[0,n] + dlon*W[1,n] + d3*W[2,n] ---------
__global__ void edge_lin3(const float* __restrict__ dll,
                          const float* __restrict__ d3,
                          const float* __restrict__ W3,
                          const float* __restrict__ bias,
                          float* __restrict__ H, int E)
{
  int i = blockIdx.x * blockDim.x + threadIdx.x;
  if (i >= E * DLAT) return;
  int e = i >> 8, n = i & 255;
  H[i] = bias[n] + dll[2 * e] * W3[n] + dll[2 * e + 1] * W3[DLAT + n]
       + d3[e] * W3[2 * DLAT + n];
}

// -------- LayerNorm over 256-wide rows (+optional fused add of 2 more
//          inputs, optional relu); one wave per row, in-place safe ---------
__global__ __launch_bounds__(256) void ln_kernel(
    const float* __restrict__ in1, const float* __restrict__ in2,
    const float* __restrict__ in3,
    const float* __restrict__ s, const float* __restrict__ o,
    float* __restrict__ out, int M, int relu)
{
  int row = blockIdx.x * 8 + (threadIdx.x >> 5);
  if (row >= M) return;
  int lane = threadIdx.x & 31;
  size_t base = (size_t)row * DLAT;
  float x[8], sum = 0.f, sq = 0.f;
#pragma unroll
  for (int j = 0; j < 8; ++j) {
    int c = lane + 32 * j;
    float v = in1[base + c];
    if (in2) v += in2[base + c];
    if (in3) v += in3[base + c];
    x[j] = v; sum += v; sq += v * v;
  }
  for (int off = 16; off > 0; off >>= 1) {
    sum += __shfl_xor(sum, off, 32);
    sq  += __shfl_xor(sq,  off, 32);
  }
  float mean = sum * (1.f / DLAT);
  float var  = sq * (1.f / DLAT) - mean * mean;
  float rstd = rsqrtf(var + 1e-5f);
#pragma unroll
  for (int j = 0; j < 8; ++j) {
    int c = lane + 32 * j;
    float v = (x[j] - mean) * rstd * s[c] + o[c];
    if (relu) v = fmaxf(v, 0.f);
    out[base + c] = v;
  }
}

// ---------------- segment-sum via scatter atomics ----------------
__global__ void scatter_add(const float* __restrict__ src,
                            const int* __restrict__ idx,
                            float* __restrict__ dst, int E)
{
  int i = blockIdx.x * blockDim.x + threadIdx.x;
  if (i >= E * DLAT) return;
  int e = i >> 8, c = i & 255;
  atomicAdd(&dst[(size_t)idx[e] * DLAT + c], src[i]);
}

// ------------- mean over 6 neighbors (linear map hoisted out) ------------
__global__ void nbr_mean(const float* __restrict__ sph,
                         const int* __restrict__ nbr,
                         float* __restrict__ out, int NS)
{
  int i = blockIdx.x * blockDim.x + threadIdx.x;
  if (i >= NS * DLAT) return;
  int sI = i >> 8, c = i & 255;
  float acc = 0.f;
#pragma unroll
  for (int j = 0; j < 6; ++j)
    acc += sph[(size_t)nbr[sI * 6 + j] * DLAT + c];
  out[i] = acc * (1.f / 6.f);
}

__global__ void fill0(float* __restrict__ p, size_t n)
{
  size_t i = (size_t)blockIdx.x * blockDim.x + threadIdx.x;
  if (i < n) p[i] = 0.f;
}

// =====================================================================
extern "C" void kernel_launch(void* const* d_in, const int* in_sizes, int n_in,
                              void* d_out, int out_size, void* d_ws, size_t ws_size,
                              hipStream_t stream)
{
  (void)in_sizes; (void)n_in; (void)out_size; (void)ws_size;
  auto f = [&](int i) { return (const float*)d_in[i]; };
  const float* features = f(0);
  const float* w_proj = f(1);   const float* b_proj = f(2);
  const float* sphere_embed = f(3);
  const float* dlatlon = f(4);
  const float* ee_w0 = f(5);    const float* ee_b0 = f(6);
  const float* ee_wr = f(7);    const float* ee_br = f(8);
  const float* ee_s = f(9);     const float* ee_o = f(10);
  const float* ee_w2 = f(11);   const float* ee_b2 = f(12);
  const float* en_w1 = f(13);   const float* en_b1 = f(14);
  const float* en_s = f(15);    const float* en_o = f(16);
  const float* en_w2 = f(17);   const float* en_b2 = f(18);
  const float* de_w0 = f(19);   const float* de_b0 = f(20);
  const float* de_wr = f(21);   const float* de_br = f(22);
  const float* de_s = f(23);    const float* de_o = f(24);
  const float* de_w2 = f(25);   const float* de_b2 = f(26);
  const float* dn_w1 = f(27);   const float* dn_b1 = f(28);
  const float* dn_s = f(29);    const float* dn_o = f(30);
  const float* dn_w2 = f(31);   const float* dn_b2 = f(32);
  const float* p_sw = f(33);    const float* p_sb = f(34);
  const float* p_nw = f(35);    const float* p_nb = f(36);
  const float* p_s = f(37);     const float* p_o = f(38);
  const float* w_out = f(39);   const float* b_out = f(40);
  const int* si  = (const int*)d_in[41];
  const int* pj  = (const int*)d_in[42];
  const int* nbr = (const int*)d_in[43];

  // ---- workspace carve-out (fp32 activations) : ~290 MB ----
  float* ws = (float*)d_ws;
  size_t off = 0;
  auto carve = [&](size_t n) { float* p = ws + off; off += n; return p; };
  float* spatialA = carve((size_t)NSP * DLAT);
  float* spatialB = carve((size_t)NSP * DLAT);
  float* sphereA  = carve((size_t)NSPH * DLAT);
  float* sphereB  = carve((size_t)NSPH * DLAT);
  float* eLat     = carve((size_t)NEDGE * DLAT);  // edge latent / nbr-mean scratch
  float* H        = carve((size_t)NSP * DLAT);    // MLP hidden (edges use prefix)
  float* msgs     = carve((size_t)NSP * DLAT);    // segment sums / processor sf,nf
  float* d3       = carve(((size_t)NEDGE + 255) & ~(size_t)255);

  auto gemm = [&](const float* A, int lda, const int* idx, const float* W, int ldw,
                  const float* bias, float* C, int ldc, int M, int N, int K, int fl) {
    dim3 grid((N + TN - 1) / TN, (M + TM - 1) / TM);
    gemm_wmma<<<grid, dim3(256), 0, stream>>>(A, lda, idx, W, ldw, bias, C, ldc, M, N, K, fl);
  };
  const int EV = NEDGE * DLAT;

  // spatial = features @ w_proj + b_proj
  gemm(features, FDIM, nullptr, w_proj, DLAT, b_proj, spatialA, DLAT, NSP, DLAT, FDIM, 0);
  // sphere = sphere_embed (working copy)
  hipMemcpyAsync(sphereA, sphere_embed, sizeof(float) * (size_t)NSPH * DLAT,
                 hipMemcpyDeviceToDevice, stream);
  // edge geometry scalar
  edge_d3<<<(NEDGE + 7) / 8, 256, 0, stream>>>(sphereA, spatialA, si, pj, d3, NEDGE);

  // ---------------- encoder: 3 message-passing steps ----------------
  for (int step = 0; step < 3; ++step) {
    if (step == 0) {
      edge_lin3<<<(EV + 255) / 256, 256, 0, stream>>>(dlatlon, d3, ee_w0, ee_b0, H, NEDGE);
      gemm(spatialA, DLAT, pj, ee_w0 + 3 * DLAT,          DLAT, nullptr, H, DLAT, NEDGE, DLAT, DLAT, 1);
      gemm(sphereA,  DLAT, si, ee_w0 + (3 + DLAT) * DLAT, DLAT, nullptr, H, DLAT, NEDGE, DLAT, DLAT, 3);
    } else {
      gemm(eLat,     DLAT, nullptr, ee_wr,                  DLAT, ee_br,   H, DLAT, NEDGE, DLAT, DLAT, 0);
      gemm(spatialA, DLAT, pj,      ee_wr + DLAT * DLAT,     DLAT, nullptr, H, DLAT, NEDGE, DLAT, DLAT, 1);
      gemm(sphereA,  DLAT, si,      ee_wr + 2 * DLAT * DLAT, DLAT, nullptr, H, DLAT, NEDGE, DLAT, DLAT, 3);
    }
    ln_kernel<<<(NEDGE + 7) / 8, 256, 0, stream>>>(H, nullptr, nullptr, ee_s, ee_o, H, NEDGE, 0);
    gemm(H, DLAT, nullptr, ee_w2, DLAT, ee_b2, eLat, DLAT, NEDGE, DLAT, DLAT, 0);
    // msgs = segment_sum(e, sphere_idx)
    fill0<<<((size_t)NSPH * DLAT + 255) / 256, 256, 0, stream>>>(msgs, (size_t)NSPH * DLAT);
    scatter_add<<<(EV + 255) / 256, 256, 0, stream>>>(eLat, si, msgs, NEDGE);
    // sphere node MLP on concat([sphere, msgs])
    gemm(sphereA, DLAT, nullptr, en_w1,               DLAT, en_b1,   H, DLAT, NSPH, DLAT, DLAT, 0);
    gemm(msgs,    DLAT, nullptr, en_w1 + DLAT * DLAT, DLAT, nullptr, H, DLAT, NSPH, DLAT, DLAT, 3);
    ln_kernel<<<(NSPH + 7) / 8, 256, 0, stream>>>(H, nullptr, nullptr, en_s, en_o, H, NSPH, 0);
    gemm(H, DLAT, nullptr, en_w2, DLAT, en_b2, sphereB, DLAT, NSPH, DLAT, DLAT, 0);
    { float* tmp = sphereA; sphereA = sphereB; sphereB = tmp; }
  }

  // ---------------- processor: 3 layers on the icosphere ----------------
  for (int l = 0; l < 3; ++l) {
    float* G1 = msgs;
    float* G2 = msgs + (size_t)NSPH * DLAT;
    nbr_mean<<<((size_t)NSPH * DLAT + 255) / 256, 256, 0, stream>>>(sphereA, nbr, eLat, NSPH);
    gemm(sphereA, DLAT, nullptr, p_sw + (size_t)l * DLAT * DLAT, DLAT, p_sb + l * DLAT, G1, DLAT, NSPH, DLAT, DLAT, 0);
    gemm(eLat,    DLAT, nullptr, p_nw + (size_t)l * DLAT * DLAT, DLAT, p_nb + l * DLAT, G2, DLAT, NSPH, DLAT, DLAT, 0);
    ln_kernel<<<(NSPH + 7) / 8, 256, 0, stream>>>(sphereA, G1, G2, p_s + l * DLAT, p_o + l * DLAT, sphereB, NSPH, 1);
    { float* tmp = sphereA; sphereA = sphereB; sphereB = tmp; }
  }

  // ---------------- decoder: 3 message-passing steps ----------------
  edge_d3<<<(NEDGE + 7) / 8, 256, 0, stream>>>(sphereA, spatialA, si, pj, d3, NEDGE);
  for (int step = 0; step < 3; ++step) {
    if (step == 0) {
      edge_lin3<<<(EV + 255) / 256, 256, 0, stream>>>(dlatlon, d3, de_w0, de_b0, H, NEDGE);
      gemm(sphereA,  DLAT, si, de_w0 + 3 * DLAT,          DLAT, nullptr, H, DLAT, NEDGE, DLAT, DLAT, 1);
      gemm(spatialA, DLAT, pj, de_w0 + (3 + DLAT) * DLAT, DLAT, nullptr, H, DLAT, NEDGE, DLAT, DLAT, 3);
    } else {
      gemm(eLat,     DLAT, nullptr, de_wr,                  DLAT, de_br,   H, DLAT, NEDGE, DLAT, DLAT, 0);
      gemm(sphereA,  DLAT, si,      de_wr + DLAT * DLAT,     DLAT, nullptr, H, DLAT, NEDGE, DLAT, DLAT, 1);
      gemm(spatialA, DLAT, pj,      de_wr + 2 * DLAT * DLAT, DLAT, nullptr, H, DLAT, NEDGE, DLAT, DLAT, 3);
    }
    ln_kernel<<<(NEDGE + 7) / 8, 256, 0, stream>>>(H, nullptr, nullptr, de_s, de_o, H, NEDGE, 0);
    gemm(H, DLAT, nullptr, de_w2, DLAT, de_b2, eLat, DLAT, NEDGE, DLAT, DLAT, 0);
    // msgs = segment_sum(e, spatial_idx) over all 65160 grid points
    fill0<<<((size_t)NSP * DLAT + 255) / 256, 256, 0, stream>>>(msgs, (size_t)NSP * DLAT);
    scatter_add<<<(EV + 255) / 256, 256, 0, stream>>>(eLat, pj, msgs, NEDGE);
    // spatial node MLP on concat([spatial, msgs])  (dominant GEMMs)
    gemm(spatialA, DLAT, nullptr, dn_w1,               DLAT, dn_b1,   H, DLAT, NSP, DLAT, DLAT, 0);
    gemm(msgs,     DLAT, nullptr, dn_w1 + DLAT * DLAT, DLAT, nullptr, H, DLAT, NSP, DLAT, DLAT, 3);
    ln_kernel<<<(NSP + 7) / 8, 256, 0, stream>>>(H, nullptr, nullptr, dn_s, dn_o, H, NSP, 0);
    gemm(H, DLAT, nullptr, dn_w2, DLAT, dn_b2, spatialB, DLAT, NSP, DLAT, DLAT, 0);
    { float* tmp = spatialA; spatialA = spatialB; spatialB = tmp; }
  }

  // ---------------- output head: 65160x256 @ 256x78 ----------------
  gemm(spatialA, DLAT, nullptr, w_out, FDIM, b_out, (float*)d_out, FDIM, NSP, FDIM, DLAT, 0);
}
